// MDAMPath_7928509628549
// MI455X (gfx1250) — matrix operand order repaired and checked by reading, not compiled, not run on hardware
//
#include <hip/hip_runtime.h>
#include <hip/hip_bf16.h>

typedef __attribute__((ext_vector_type(16))) _Float16 v16h;
typedef __attribute__((ext_vector_type(8)))  float    v8f;

#define Bn 128
#define Nn 2000
#define Dn 256
#define Hn 8
#define HDn 32
#define NEGV (-1e9f)

// ---------------- K1: graph_embed = mean_n h_embed ----------------
__global__ __launch_bounds__(1024) void k_mean(const float* __restrict__ he,
                                               float* __restrict__ ge) {
    __shared__ float part[4][Dn];
    const int b = blockIdx.x;
    const int d = threadIdx.x & 255;
    const int qy = threadIdx.x >> 8;
    const float* p = he + ((size_t)b * Nn + qy * 500) * Dn + d;
    float acc = 0.f;
    for (int n = 0; n < 500; ++n) acc += p[(size_t)n * Dn];
    part[qy][d] = acc;
    __syncthreads();
    if (qy == 0)
        ge[b * Dn + d] = (part[0][d] + part[1][d] + part[2][d] + part[3][d]) * (1.0f / (float)Nn);
}

// ------- K2: q = ge@Wf + [h_first,h_cur]@Ws ; wqB pre-swizzled f16 B-fragments -------
// wqB layout: [b][kstep(8)][lane(32)][e(16)] halves, where for lane l = kh*16 + h,
// element e holds wq[d = kstep*32 + kh*16 + e][h]  (h>=8 columns are zero padding).
__global__ __launch_bounds__(256) void k_qwq(const float* __restrict__ he,
                                             const float* __restrict__ ge,
                                             const float* __restrict__ Wf,
                                             const float* __restrict__ Ws,
                                             const float* __restrict__ Wn,
                                             const int* __restrict__ firstn,
                                             const int* __restrict__ curn,
                                             float* __restrict__ q,
                                             _Float16* __restrict__ wqB) {
    __shared__ float qs[Dn];
    const int b = blockIdx.x, d = threadIdx.x;
    const int f = firstn[b], c = curn[b];
    const float* g  = ge + b * Dn;
    const float* hf = he + ((size_t)(b * Nn + f)) * Dn;
    const float* hc = he + ((size_t)(b * Nn + c)) * Dn;
    float acc = 0.f;
    for (int e = 0; e < Dn; ++e)
        acc += g[e] * Wf[e * Dn + d] + hf[e] * Ws[e * Dn + d] + hc[e] * Ws[(Dn + e) * Dn + d];
    q[b * Dn + d] = acc;
    qs[d] = acc;
    __syncthreads();
    // this thread owns matrix row d of wq; scatter into swizzled fragment layout
    const int kstep = d >> 5;
    const int kh    = (d >> 4) & 1;
    const int e     = d & 15;
    _Float16* wbase = wqB + (((size_t)(b * 8 + kstep)) * 32 + kh * 16) * 16 + e;
    for (int h = 0; h < 16; ++h) {
        float a = 0.f;
        if (h < Hn) {
            const float* wr = Wn + (size_t)d * (3 * Dn) + h * HDn;   // Wk slice
            for (int j = 0; j < HDn; ++j) a += wr[j] * qs[h * HDn + j];
        }
        wbase[(size_t)h * 16] = (_Float16)a;
    }
}

// ---------------- K3: scores via WMMA (16 n-rows x 16 head-cols, K=256) ----------------
__global__ __launch_bounds__(256) void k_scores(const float* __restrict__ he,
                                                const float* __restrict__ gk,
                                                const _Float16* __restrict__ wqB,
                                                const float* __restrict__ q,
                                                const unsigned char* __restrict__ mask,
                                                float* __restrict__ S) {
    const int b    = blockIdx.x;
    const int wid  = threadIdx.x >> 5;
    const int lane = threadIdx.x & 31;
    const int tile = blockIdx.y * 8 + wid;
    if (tile >= Nn / 16) return;                  // 125 tiles, wave-uniform
    const int n0 = tile * 16;
    const int m  = lane & 15;                     // A row within tile / C column (head)
    const int kh = lane >> 4;                     // lane half

    const float* arow = he + ((size_t)(b * Nn + n0 + m)) * Dn + kh * 16;
    const v16h* bfrag = (const v16h*)(wqB + (((size_t)b * 8) * 32 + lane) * 16);

    v8f c = {};
    #pragma unroll
    for (int k0 = 0; k0 < Dn; k0 += 32) {
        // A fragment: lane holds K = k0+kh*16+{0..15}
        const float4* ap = (const float4*)(arow + k0);
        float4 a0 = ap[0], a1 = ap[1], a2 = ap[2], a3 = ap[3];
        v16h a;
        a[0]=(_Float16)a0.x; a[1]=(_Float16)a0.y; a[2]=(_Float16)a0.z; a[3]=(_Float16)a0.w;
        a[4]=(_Float16)a1.x; a[5]=(_Float16)a1.y; a[6]=(_Float16)a1.z; a[7]=(_Float16)a1.w;
        a[8]=(_Float16)a2.x; a[9]=(_Float16)a2.y; a[10]=(_Float16)a2.z; a[11]=(_Float16)a2.w;
        a[12]=(_Float16)a3.x; a[13]=(_Float16)a3.y; a[14]=(_Float16)a3.z; a[15]=(_Float16)a3.w;
        // B fragment: one contiguous 32B vector load per lane per step
        v16h bv = bfrag[(k0 >> 5) * 32];
        c = __builtin_amdgcn_wmma_f32_16x16x32_f16(false, a, false, bv, (short)0, c, false, false);
    }

    // Epilogue: lane column = head h (only h<8 valid); rows = kh*8 + r
    const int h = m;
    if (h < Hn) {
        const float4* q4 = (const float4*)(q + b * Dn + h * HDn);
        float qreg[HDn];
        #pragma unroll
        for (int j = 0; j < 8; ++j) {
            float4 t = q4[j];
            qreg[4*j] = t.x; qreg[4*j+1] = t.y; qreg[4*j+2] = t.z; qreg[4*j+3] = t.w;
        }
        #pragma unroll
        for (int r = 0; r < 8; ++r) {
            const int n = n0 + kh * 8 + r;
            const float4* gp = (const float4*)(gk + ((size_t)(h * Bn + b) * Nn + n) * HDn);
            float acc = 0.f;
            #pragma unroll
            for (int j = 0; j < 8; ++j) {
                float4 t = gp[j];
                acc += qreg[4*j]*t.x + qreg[4*j+1]*t.y + qreg[4*j+2]*t.z + qreg[4*j+3]*t.w;
            }
            float s = (c[r] + acc) * 0.17677669529663687f;     // 1/sqrt(32)
            if (!mask[b * Nn + n]) s = NEGV;
            S[((size_t)(b * Hn + h)) * Nn + n] = s;
        }
    }
}

// ---------------- K4: softmax over n per (b,h); ctx = attn@he ; cv = attn@gv_dyn ----------------
__global__ __launch_bounds__(256) void k_attn(const float* __restrict__ he,
                                              const float* __restrict__ gv,
                                              const float* __restrict__ S,
                                              float* __restrict__ ctx,
                                              float* __restrict__ cv) {
    extern __shared__ float smem[];               // [8*2000] attn + [256] red
    float* attn = smem;
    float* red  = smem + Hn * Nn;
    const int b = blockIdx.x, t = threadIdx.x;

    for (int h = 0; h < Hn; ++h) {
        const float* Sr = S + ((size_t)(b * Hn + h)) * Nn;
        float lm = -3.4e38f;
        for (int n = t; n < Nn; n += 256) { float v = Sr[n]; attn[h * Nn + n] = v; lm = fmaxf(lm, v); }
        red[t] = lm; __syncthreads();
        for (int s = 128; s > 0; s >>= 1) { if (t < s) red[t] = fmaxf(red[t], red[t + s]); __syncthreads(); }
        const float M = red[0]; __syncthreads();
        float ls = 0.f;
        for (int n = t; n < Nn; n += 256) { float e = expf(attn[h * Nn + n] - M); attn[h * Nn + n] = e; ls += e; }
        red[t] = ls; __syncthreads();
        for (int s = 128; s > 0; s >>= 1) { if (t < s) red[t] += red[t + s]; __syncthreads(); }
        const float inv = 1.0f / red[0]; __syncthreads();
        for (int n = t; n < Nn; n += 256) attn[h * Nn + n] *= inv;
        __syncthreads();
    }
    // ctx[b][h][d], thread t = d, 8 FMA per loaded element (single pass over h_embed[b])
    float acc[Hn] = {};
    const float* hb = he + (size_t)b * Nn * Dn + t;
    for (int n = 0; n < Nn; ++n) {
        const float x = hb[(size_t)n * Dn];
        #pragma unroll
        for (int h = 0; h < Hn; ++h) acc[h] += attn[h * Nn + n] * x;
    }
    for (int h = 0; h < Hn; ++h) ctx[((size_t)(b * Hn + h)) * Dn + t] = acc[h];
    // cv[b][h][j], thread t -> (h=t>>5, j=t&31)
    const int h = t >> 5, j = t & 31;
    const float* gp = gv + ((size_t)(h * Bn + b)) * Nn * HDn + j;
    float a2 = 0.f;
    for (int n = 0; n < Nn; ++n) a2 += attn[h * Nn + n] * gp[(size_t)n * HDn];
    cv[b * Dn + t] = a2;
}

// ---------------- K5: heads -> glimpse -> v = W_l @ glimpse ----------------
__global__ __launch_bounds__(256) void k_head(const float* __restrict__ ctx,
                                              const float* __restrict__ cv,
                                              const float* __restrict__ Wn,
                                              const float* __restrict__ Wout,
                                              float* __restrict__ glimpse,
                                              float* __restrict__ v) {
    __shared__ float hd[Dn], gl[Dn];
    const int b = blockIdx.x, t = threadIdx.x;
    const int h = t >> 5, j = t & 31;
    const float* cr = ctx + ((size_t)(b * Hn + h)) * Dn;
    float a = 0.f;
    for (int d = 0; d < Dn; ++d) a += cr[d] * Wn[(size_t)d * (3 * Dn) + Dn + h * HDn + j];
    hd[t] = a + cv[b * Dn + t];
    __syncthreads();
    float ag = 0.f;
    for (int c2 = 0; c2 < Dn; ++c2) ag += hd[c2] * Wout[(size_t)c2 * Dn + t];
    glimpse[b * Dn + t] = ag; gl[t] = ag;
    __syncthreads();
    float av = 0.f;
    const float* wl = Wn + (size_t)t * (3 * Dn) + 2 * Dn;
    for (int e = 0; e < Dn; ++e) av += wl[e] * gl[e];
    v[b * Dn + t] = av;
}

// ---------------- K6: logits = tanh((he.v + g.lk)/16)*10, mask, log_softmax ----------------
__global__ __launch_bounds__(256) void k_logits(const float* __restrict__ he,
                                                const float* __restrict__ lk,
                                                const float* __restrict__ v,
                                                const float* __restrict__ g,
                                                const unsigned char* __restrict__ mask,
                                                float* __restrict__ out) {
    __shared__ float vv[Dn], gg[Dn], red[256];
    __shared__ float lg[Nn];
    const int b = blockIdx.x, t = threadIdx.x;
    vv[t] = v[b * Dn + t]; gg[t] = g[b * Dn + t];
    __syncthreads();
    const int w = t >> 5, lane = t & 31;
    for (int n = w; n < Nn; n += 8) {
        const float* hp = he + ((size_t)(b * Nn + n)) * Dn;
        const float* lp = lk + ((size_t)(b * Nn + n)) * Dn;
        float p = 0.f;
        for (int e = lane; e < Dn; e += 32) p += hp[e] * vv[e] + lp[e] * gg[e];
        for (int off = 16; off > 0; off >>= 1) p += __shfl_xor(p, off, 32);
        if (lane == 0) {
            float s = tanhf(p * 0.0625f) * 10.0f;   // /sqrt(256), tanh clip
            if (!mask[b * Nn + n]) s = NEGV;
            lg[n] = s;
        }
    }
    __syncthreads();
    float lm = -3.4e38f;
    for (int n = t; n < Nn; n += 256) lm = fmaxf(lm, lg[n]);
    red[t] = lm; __syncthreads();
    for (int s = 128; s > 0; s >>= 1) { if (t < s) red[t] = fmaxf(red[t], red[t + s]); __syncthreads(); }
    const float M = red[0]; __syncthreads();
    float ls = 0.f;
    for (int n = t; n < Nn; n += 256) ls += expf(lg[n] - M);
    red[t] = ls; __syncthreads();
    for (int s = 128; s > 0; s >>= 1) { if (t < s) red[t] += red[t + s]; __syncthreads(); }
    const float L = M + logf(red[0]);
    for (int n = t; n < Nn; n += 256) out[(size_t)b * Nn + n] = lg[n] - L;
}

extern "C" void kernel_launch(void* const* d_in, const int* in_sizes, int n_in,
                              void* d_out, int out_size, void* d_ws, size_t ws_size,
                              hipStream_t stream) {
    const float* he = (const float*)d_in[0];
    const float* gk = (const float*)d_in[1];
    const float* gv = (const float*)d_in[2];
    const float* lk = (const float*)d_in[3];
    const float* Wn = (const float*)d_in[4];
    const float* Wf = (const float*)d_in[5];
    const float* Ws = (const float*)d_in[6];
    const float* Wo = (const float*)d_in[7];
    const int* fn = (const int*)d_in[8];
    const int* cn = (const int*)d_in[9];
    const unsigned char* mask = (const unsigned char*)d_in[10];
    float* out = (float*)d_out;

    char* ws = (char*)d_ws;
    float* ge      = (float*)(ws + 0);          // 128*256 f32
    float* q       = (float*)(ws + 131072);     // 128*256 f32
    float* S       = (float*)(ws + 262144);     // 128*8*2000 f32
    float* ctx     = (float*)(ws + 8454144);    // 128*8*256 f32
    float* cv      = (float*)(ws + 9502720);    // 128*256 f32
    float* glimpse = (float*)(ws + 9633792);    // 128*256 f32
    float* v       = (float*)(ws + 9764864);    // 128*256 f32
    _Float16* wqB  = (_Float16*)(ws + 9895936); // 128*8*32*16 f16 (swizzled B fragments)

    k_mean  <<<Bn, 1024, 0, stream>>>(he, ge);
    k_qwq   <<<Bn, 256, 0, stream>>>(he, ge, Wf, Ws, Wn, fn, cn, q, wqB);
    k_scores<<<dim3(Bn, 16), 256, 0, stream>>>(he, gk, wqB, q, mask, S);
    k_attn  <<<Bn, 256, (Hn * Nn + 256) * sizeof(float), stream>>>(he, gv, S, ctx, cv);
    k_head  <<<Bn, 256, 0, stream>>>(ctx, cv, Wn, Wo, glimpse, v);
    k_logits<<<Bn, 256, 0, stream>>>(he, lk, v, glimpse, mask, out);
}
